// DEQGCN_67095979098485
// MI455X (gfx1250) — matrix operand order repaired and checked by reading, not compiled, not run on hardware
//
#include <hip/hip_runtime.h>
#include <hip/hip_bf16.h>

typedef __attribute__((ext_vector_type(16))) __bf16 v16bf;
typedef __attribute__((ext_vector_type(8)))  float  v8f;
typedef __attribute__((ext_vector_type(4)))  float  v4f;
typedef __attribute__((ext_vector_type(2)))  float  v2f;

#define N_NODES 100000
#define NEDGE   1200000
#define FIN     512
#define HID     64
#define NCLS    40
#define MAXITER 32
#define LN_EPS  1e-5f

// ---------------------------------------------------------------------------
// A-fragment loader: 16x32 bf16 A tile for v_wmma_f32_16x16x32_bf16, built
// from a row-major f32 matrix. Layout per ISA 7.12.2 (16-bit A 16x32):
//   lane L: m = L&15, kbase = (L<16)?0:8
//   elem e<8  -> K = kbase + e
//   elem e>=8 -> K = kbase + 8 + e   (i.e. kbase+16 .. kbase+23)
// ---------------------------------------------------------------------------
__device__ inline v16bf load_a_frag(const float* src, int row_stride,
                                    int rowbase, int kofs, int lane) {
  int m  = lane & 15;
  int kb = (lane < 16) ? 0 : 8;
  const float* p = src + (size_t)(rowbase + m) * row_stride + kofs + kb;
  v4f a0 = *(const v4f*)(p);
  v4f a1 = *(const v4f*)(p + 4);
  v4f a2 = *(const v4f*)(p + 16);
  v4f a3 = *(const v4f*)(p + 20);
  v16bf r;
#pragma unroll
  for (int i = 0; i < 4; i++) {
    r[i]      = (__bf16)a0[i];
    r[4 + i]  = (__bf16)a1[i];
    r[8 + i]  = (__bf16)a2[i];
    r[12 + i] = (__bf16)a3[i];
  }
  return r;
}

// ---------------------------------------------------------------------------
// Convert a row-major f32 weight [K x ncols_src] into bf16 B fragments.
// B layout (32x16 K x N tile): lane L covers column n = t*16 + (L&15);
// kbase = (L<16)?0:8; elem e<8 -> K = kbase+e, e>=8 -> K = kbase+8+e.
// One block (32 threads) per (kb, t) tile. Pads n >= ncols_src with 0.
// ---------------------------------------------------------------------------
__global__ __launch_bounds__(32) void k_cvt_b(const float* W, v16bf* dst,
                                              int ncols_src, int T) {
  int kb = blockIdx.x / T, t = blockIdx.x % T;
  int lane = threadIdx.x;
  int n = t * 16 + (lane & 15);
  int kbase = kb * 32 + ((lane < 16) ? 0 : 8);
  v16bf r;
#pragma unroll
  for (int e = 0; e < 16; e++) {
    float v = (n < ncols_src) ? W[(size_t)(kbase + e + (e >= 8 ? 8 : 0)) * ncols_src + n] : 0.0f;
    r[e] = (__bf16)v;
  }
  dst[(size_t)blockIdx.x * 32 + lane] = r;
}

// ------------------------- CSR build (one-time) ----------------------------
__global__ void k_zero_i32(int* p, int n) {
  int i = blockIdx.x * blockDim.x + threadIdx.x;
  if (i < n) p[i] = 0;
}
__global__ void k_hist(const int* row, int* deg) {
  int i = blockIdx.x * blockDim.x + threadIdx.x;
  if (i < NEDGE) atomicAdd(&deg[row[i]], 1);
}
__global__ __launch_bounds__(1024) void k_scan_block(const int* deg, int* rs,
                                                     int* part) {
  __shared__ int sh[1024];
  int gid = blockIdx.x * 1024 + threadIdx.x;
  int v = (gid < N_NODES) ? deg[gid] : 0;
  sh[threadIdx.x] = v;
  __syncthreads();
  for (int off = 1; off < 1024; off <<= 1) {
    int t = (threadIdx.x >= off) ? sh[threadIdx.x - off] : 0;
    __syncthreads();
    sh[threadIdx.x] += t;
    __syncthreads();
  }
  if (gid < N_NODES) rs[gid] = sh[threadIdx.x] - v;  // exclusive within block
  if (threadIdx.x == 1023) part[blockIdx.x] = sh[1023];
}
__global__ void k_scan_part(int* part, int nblk, int* rs) {
  if (blockIdx.x == 0 && threadIdx.x == 0) {
    int run = 0;
    for (int b = 0; b < nblk; b++) { int t = part[b]; part[b] = run; run += t; }
    rs[N_NODES] = run;  // == NEDGE
  }
}
__global__ void k_scan_add(int* rs, const int* part, int* cur) {
  int i = blockIdx.x * blockDim.x + threadIdx.x;
  if (i < N_NODES) {
    int v = rs[i] + part[i >> 10];
    rs[i] = v;
    cur[i] = v;
  }
}
__global__ void k_fill(const int* row, const int* col, const float* data,
                       int* cur, int* col_s, float* w_s) {
  int i = blockIdx.x * blockDim.x + threadIdx.x;
  if (i < NEDGE) {
    int p = atomicAdd(&cur[row[i]], 1);
    col_s[p] = col[i];
    w_s[p] = data[i];
  }
}

// ------------- x = LN1(relu(nf @ W1 + b1)), z = 0  (WMMA bf16) -------------
__global__ __launch_bounds__(256) void k_input(const float* nf, const float* b1,
                                               const float* sc, const float* of,
                                               const v16bf* W1bf, float* x,
                                               float* z) {
  int wave = blockIdx.x * (blockDim.x >> 5) + (threadIdx.x >> 5);
  if (wave >= N_NODES / 16) return;
  int lane = threadIdx.x & 31;
  int rowbase = wave * 16;
  v8f c[4] = {};
  for (int ks = 0; ks < FIN / 32; ks++) {
    v16bf a = load_a_frag(nf, FIN, rowbase, ks * 32, lane);
#pragma unroll
    for (int t = 0; t < 4; t++) {
      v16bf b = W1bf[(size_t)(ks * 4 + t) * 32 + lane];
      c[t] = __builtin_amdgcn_wmma_f32_16x16x32_bf16(false, a, false, b,
                                                     (short)0, c[t], false, false);
    }
  }
  int n15 = lane & 15;
  int mofs = (lane < 16) ? 0 : 8;
  float b1v[4], scv[4], ofv[4];
#pragma unroll
  for (int t = 0; t < 4; t++) {
    int n = t * 16 + n15;
    b1v[t] = b1[n]; scv[t] = sc[n]; ofv[t] = of[n];
  }
#pragma unroll
  for (int r = 0; r < 8; r++) {
    float vals[4], s1 = 0.f, s2 = 0.f;
#pragma unroll
    for (int t = 0; t < 4; t++) {
      float v = c[t][r] + b1v[t];
      v = v > 0.f ? v : 0.f;
      vals[t] = v; s1 += v; s2 += v * v;
    }
#pragma unroll
    for (int m = 1; m < 16; m <<= 1) {  // reduce within 16-lane halves
      s1 += __shfl_xor(s1, m, 32);
      s2 += __shfl_xor(s2, m, 32);
    }
    float mean = s1 * (1.f / HID);
    float var = s2 * (1.f / HID) - mean * mean;
    float inv = rsqrtf(var + LN_EPS);
    size_t rowb = (size_t)(rowbase + mofs + r) * HID;
#pragma unroll
    for (int t = 0; t < 4; t++) {
      x[rowb + t * 16 + n15] = (vals[t] - mean) * inv * scv[t] + ofv[t];
      z[rowb + t * 16 + n15] = 0.0f;  // z0 = 0
    }
  }
}

// ---------------------- zz = z @ Wg  (WMMA bf16) ---------------------------
__global__ __launch_bounds__(256) void k_gemm_hid(const float* zin,
                                                  const v16bf* Bbf,
                                                  float* zout) {
  int wave = blockIdx.x * (blockDim.x >> 5) + (threadIdx.x >> 5);
  if (wave >= N_NODES / 16) return;
  int lane = threadIdx.x & 31;
  int rowbase = wave * 16;
  v8f c[4] = {};
#pragma unroll
  for (int ks = 0; ks < 2; ks++) {
    v16bf a = load_a_frag(zin, HID, rowbase, ks * 32, lane);
#pragma unroll
    for (int t = 0; t < 4; t++) {
      v16bf b = Bbf[(size_t)(ks * 4 + t) * 32 + lane];
      c[t] = __builtin_amdgcn_wmma_f32_16x16x32_bf16(false, a, false, b,
                                                     (short)0, c[t], false, false);
    }
  }
  int n15 = lane & 15;
  int mofs = (lane < 16) ? 0 : 8;
#pragma unroll
  for (int r = 0; r < 8; r++) {
    size_t rowb = (size_t)(rowbase + mofs + r) * HID;
#pragma unroll
    for (int t = 0; t < 4; t++) zout[rowb + t * 16 + n15] = c[t][r];
  }
}

// --------- z = LN2(relu(CSR-SpMM(zz) + x)) : one wave per row --------------
// Lane L owns feature pair (2L, 2L+1) -> single b64 gather per edge per lane.
__global__ __launch_bounds__(256) void k_spmm_ln(const int* rs, const int* col_s,
                                                 const float* w_s,
                                                 const float* zz, const float* x,
                                                 const float* sc, const float* of,
                                                 float* z) {
  int wave = blockIdx.x * (blockDim.x >> 5) + (threadIdx.x >> 5);
  if (wave >= N_NODES) return;
  int lane = threadIdx.x & 31;
  int f = lane * 2;
  int s = rs[wave], e = rs[wave + 1];
  float a0 = 0.f, a1 = 0.f;
  int i = s;
  // 2x unrolled edge loop: two independent b64 gathers in flight
  for (; i + 1 < e; i += 2) {
    int c0 = col_s[i];     float w0 = w_s[i];
    int c1 = col_s[i + 1]; float w1 = w_s[i + 1];
    v2f p0 = *(const v2f*)(zz + (size_t)c0 * HID + f);
    v2f p1 = *(const v2f*)(zz + (size_t)c1 * HID + f);
    a0 += w0 * p0.x; a1 += w0 * p0.y;
    a0 += w1 * p1.x; a1 += w1 * p1.y;
  }
  if (i < e) {
    int c0 = col_s[i]; float w0 = w_s[i];
    v2f p0 = *(const v2f*)(zz + (size_t)c0 * HID + f);
    a0 += w0 * p0.x; a1 += w0 * p0.y;
  }
  size_t base = (size_t)wave * HID;
  v2f xr = *(const v2f*)(x + base + f);
  float v0 = a0 + xr.x; v0 = v0 > 0.f ? v0 : 0.f;
  float v1 = a1 + xr.y; v1 = v1 > 0.f ? v1 : 0.f;
  float s1 = v0 + v1, s2 = v0 * v0 + v1 * v1;
#pragma unroll
  for (int m = 1; m < 32; m <<= 1) {
    s1 += __shfl_xor(s1, m, 32);
    s2 += __shfl_xor(s2, m, 32);
  }
  float mean = s1 * (1.f / HID);
  float var = s2 * (1.f / HID) - mean * mean;
  float inv = rsqrtf(var + LN_EPS);
  v2f scv = *(const v2f*)(sc + f);
  v2f ofv = *(const v2f*)(of + f);
  v2f zo;
  zo.x = (v0 - mean) * inv * scv.x + ofv.x;
  zo.y = (v1 - mean) * inv * scv.y + ofv.y;
  *(v2f*)(z + base + f) = zo;
}

// -------------------- out = z @ W2 + b2  (WMMA bf16) -----------------------
__global__ __launch_bounds__(256) void k_out(const float* z, const v16bf* W2bf,
                                             const float* b2, float* out) {
  int wave = blockIdx.x * (blockDim.x >> 5) + (threadIdx.x >> 5);
  if (wave >= N_NODES / 16) return;
  int lane = threadIdx.x & 31;
  int rowbase = wave * 16;
  v8f c[3] = {};
#pragma unroll
  for (int ks = 0; ks < 2; ks++) {
    v16bf a = load_a_frag(z, HID, rowbase, ks * 32, lane);
#pragma unroll
    for (int t = 0; t < 3; t++) {
      v16bf b = W2bf[(size_t)(ks * 3 + t) * 32 + lane];
      c[t] = __builtin_amdgcn_wmma_f32_16x16x32_bf16(false, a, false, b,
                                                     (short)0, c[t], false, false);
    }
  }
  int n15 = lane & 15;
  int mofs = (lane < 16) ? 0 : 8;
#pragma unroll
  for (int t = 0; t < 3; t++) {
    int n = t * 16 + n15;
    if (n < NCLS) {
      float bb = b2[n];
#pragma unroll
      for (int r = 0; r < 8; r++)
        out[(size_t)(rowbase + mofs + r) * NCLS + n] = c[t][r] + bb;
    }
  }
}

// ---------------------------------------------------------------------------
extern "C" void kernel_launch(void* const* d_in, const int* in_sizes, int n_in,
                              void* d_out, int out_size, void* d_ws,
                              size_t ws_size, hipStream_t stream) {
  (void)in_sizes; (void)n_in; (void)out_size; (void)ws_size;
  const float* nf   = (const float*)d_in[0];
  const float* data = (const float*)d_in[1];
  const int*   row  = (const int*)d_in[2];
  const int*   col  = (const int*)d_in[3];
  const float* W1   = (const float*)d_in[4];
  const float* b1   = (const float*)d_in[5];
  const float* ln1s = (const float*)d_in[6];
  const float* ln1o = (const float*)d_in[7];
  const float* Wg   = (const float*)d_in[8];
  const float* ln2s = (const float*)d_in[9];
  const float* ln2o = (const float*)d_in[10];
  const float* W2   = (const float*)d_in[11];
  const float* b2   = (const float*)d_in[12];
  float* out = (float*)d_out;

  char* ws = (char*)d_ws;
  auto alloc = [&](size_t bytes) -> char* {
    char* p = ws;
    ws += (bytes + 255) & ~(size_t)255;
    return p;
  };
  const size_t NH = (size_t)N_NODES * HID;
  float* x     = (float*)alloc(NH * sizeof(float));
  float* z     = (float*)alloc(NH * sizeof(float));
  float* zz    = (float*)alloc(NH * sizeof(float));
  int*   deg   = (int*)alloc((size_t)N_NODES * 4);
  int*   rs    = (int*)alloc((size_t)(N_NODES + 1) * 4);
  int*   cur   = (int*)alloc((size_t)N_NODES * 4);
  int*   part  = (int*)alloc(1024);
  int*   col_s = (int*)alloc((size_t)NEDGE * 4);
  float* w_s   = (float*)alloc((size_t)NEDGE * 4);
  v16bf* W1bf  = (v16bf*)alloc(64 * 32 * sizeof(v16bf));   // KB=16, T=4
  v16bf* Wgbf  = (v16bf*)alloc(8 * 32 * sizeof(v16bf));    // KB=2,  T=4
  v16bf* W2bf  = (v16bf*)alloc(6 * 32 * sizeof(v16bf));    // KB=2,  T=3

  const int nblk_scan = (N_NODES + 1023) / 1024;           // 98
  const int gN  = (N_NODES + 255) / 256;
  const int gE  = (NEDGE + 255) / 256;
  const int gW  = (N_NODES / 16 + 7) / 8;                  // 16-row wave tiles
  const int gR  = (N_NODES + 7) / 8;                       // 1 row per wave

  // One-time weight conversion to WMMA bf16 B-fragment layout
  k_cvt_b<<<64, 32, 0, stream>>>(W1, W1bf, HID, 4);
  k_cvt_b<<<8, 32, 0, stream>>>(Wg, Wgbf, HID, 4);
  k_cvt_b<<<6, 32, 0, stream>>>(W2, W2bf, NCLS, 3);

  // One-time CSR build (graph is static across all 32 iterations)
  k_zero_i32<<<gN, 256, 0, stream>>>(deg, N_NODES);
  k_hist<<<gE, 256, 0, stream>>>(row, deg);
  k_scan_block<<<nblk_scan, 1024, 0, stream>>>(deg, rs, part);
  k_scan_part<<<1, 1, 0, stream>>>(part, nblk_scan, rs);
  k_scan_add<<<gN, 256, 0, stream>>>(rs, part, cur);
  k_fill<<<gE, 256, 0, stream>>>(row, col, data, cur, col_s, w_s);

  // x = LN1(relu(nf @ W1 + b1)); z = 0
  k_input<<<gW, 256, 0, stream>>>(nf, b1, ln1s, ln1o, W1bf, x, z);

  // DEQ fixed-point loop (all traffic L2-resident)
  for (int it = 0; it < MAXITER; ++it) {
    k_gemm_hid<<<gW, 256, 0, stream>>>(z, Wgbf, zz);
    k_spmm_ln<<<gR, 256, 0, stream>>>(rs, col_s, w_s, zz, x, ln2s, ln2o, z);
  }

  // out = z @ W2 + b2
  k_out<<<gW, 256, 0, stream>>>(z, W2bf, b2, out);
}
